// Policy_86294482911517
// MI455X (gfx1250) — compile-verified, matrix operand order checked by reading, not run on hardware
//
#include <hip/hip_runtime.h>
#include <hip/hip_bf16.h>

// ---------------- problem constants ----------------
static constexpr int KNLEAVES = 4096;
static constexpr int KNF      = 4096;              // leaf feature dim
static constexpr int KNN      = 2 * KNLEAVES - 1;  // 8191 tree nodes
static constexpr int KROWS    = KNN + 1;           // 8192 (with focal row)
static constexpr int KHID     = 512;
static constexpr int KE       = 4 * KHID + 2;      // 2050
static constexpr int KEP      = 2080;              // 2050 padded to %32

typedef __bf16 bf16_t;
typedef __attribute__((ext_vector_type(16))) __bf16 v16bf;
typedef __attribute__((ext_vector_type(8)))  __bf16 v8bf;
typedef __attribute__((ext_vector_type(8)))  float  v8f;

// ---------------- c-value recursion (inherently serial, 8191 steps) --------
__global__ void k_cvals(const int* __restrict__ ns, const int* __restrict__ ch0,
                        const int* __restrict__ ch1, float* __restrict__ c) {
  if (threadIdx.x == 0 && blockIdx.x == 0) {
    for (int u = 0; u < KNN; ++u) {
      if (ns[u] >= 0) c[u] = 0.0f;                     // leaf: masked to 0
      else            c[u] = 1.0f / (3.0f - c[ch0[u]] - c[ch1[u]]);
    }
  }
}

// ---------------- belief prop: per-wave 32-column slab ---------------------
// d has KROWS rows (row KNN = focal one-hot), KNF cols, f32 row-major.
__global__ __launch_bounds__(32) void k_belief(
    const int* __restrict__ ns, const int* __restrict__ ch0,
    const int* __restrict__ ch1, const int* __restrict__ parent,
    const float* __restrict__ c, const int* __restrict__ focal,
    float* __restrict__ d) {
  const int col = blockIdx.x * 32 + threadIdx.x;
  const int f   = focal[0];
  // leaf rows: one-hot(node_sample)
  for (int u = 0; u < KNLEAVES; ++u)
    d[(size_t)u * KNF + col] = (ns[u] == col) ? 1.0f : 0.0f;
  // focal row
  d[(size_t)KNN * KNF + col] = (col == f) ? 1.0f : 0.0f;
  // postorder: internal nodes (children have smaller ids)
  for (int u = KNLEAVES; u < KNN; ++u) {
    float dd = d[(size_t)ch0[u] * KNF + col] + d[(size_t)ch1[u] * KNF + col];
    d[(size_t)u * KNF + col] = c[u] * dd;
  }
  // preorder correction (root unchanged; leaves have c==0 -> unchanged)
  for (int u = KNN - 2; u >= KNLEAVES; --u)
    d[(size_t)u * KNF + col] += c[u] * d[(size_t)parent[u] * KNF + col];
}

// ---------------- GCN aggregation: mean over {self} U valid neighbors ------
__global__ void k_agg(const float* __restrict__ x, const int* __restrict__ neigh,
                      bf16_t* __restrict__ out, int width) {
  const int col = blockIdx.x * blockDim.x + threadIdx.x;
  const int u   = blockIdx.y;
  if (col >= width) return;
  float s = x[(size_t)u * width + col];
  float deg = 1.0f;
  #pragma unroll
  for (int j = 0; j < 3; ++j) {
    int v = neigh[u * 3 + j];
    if (v >= 0) { s += x[(size_t)v * width + col]; deg += 1.0f; }
  }
  out[(size_t)u * width + col] = (bf16_t)(s / deg);
}

// ---------------- weight prep: f32 (K x N) -> bf16 transposed (N x Kpad) ---
__global__ void k_wt(const float* __restrict__ W, bf16_t* __restrict__ Bt,
                     int K, int N, int Kpad) {
  const int k = blockIdx.x * blockDim.x + threadIdx.x;
  const int n = blockIdx.y;
  if (k >= Kpad) return;
  Bt[(size_t)n * Kpad + k] = (k < K) ? (bf16_t)W[(size_t)k * N + n] : (bf16_t)0.0f;
}

// ---------------- fragment loader: two contiguous 16B chunks ---------------
__device__ __forceinline__ v16bf load_frag(const bf16_t* row, int kb) {
  v8bf lo = *(const v8bf*)(row + kb);
  v8bf hi = *(const v8bf*)(row + kb + 16);
  return __builtin_shufflevector(lo, hi, 0,1,2,3,4,5,6,7,8,9,10,11,12,13,14,15);
}

// ---------------- WMMA bf16 GEMM: C(MxN) = act(A(MxK) @ Bt(NxK)^T + bias) --
// block = 4 waves; wave -> 32x64 tile (2 A frags x 4 B frags -> 8 WMMA/K-step)
__global__ __launch_bounds__(128) void k_gemm_wmma(
    const bf16_t* __restrict__ A, const bf16_t* __restrict__ Bt,
    const float* __restrict__ bias, float* __restrict__ C,
    bf16_t* __restrict__ Cbf, int N, int K, int act) {
  const int lane = threadIdx.x & 31;
  const int wave = threadIdx.x >> 5;
  const int m0 = (blockIdx.x * 4 + wave) * 32;
  const int n0 = blockIdx.y * 64;
  const int hl  = lane >> 4;   // half-wave select
  const int l16 = lane & 15;

  const bf16_t* arow0 = A + (size_t)(m0 + l16) * K;
  const bf16_t* arow1 = A + (size_t)(m0 + 16 + l16) * K;
  const bf16_t* brow0 = Bt + (size_t)(n0 +  0 + l16) * K;
  const bf16_t* brow1 = Bt + (size_t)(n0 + 16 + l16) * K;
  const bf16_t* brow2 = Bt + (size_t)(n0 + 32 + l16) * K;
  const bf16_t* brow3 = Bt + (size_t)(n0 + 48 + l16) * K;

  v8f acc[8] = {};  // [i*4+s]: i = M-subtile (0..1), s = N-subtile (0..3)

  for (int k0 = 0; k0 < K; k0 += 32) {
    const int kb = k0 + hl * 8;  // lanes<16: K {0-7,16-23}; lanes>=16: {8-15,24-31}
    // speculative prefetch one K-step ahead (dropped past end of buffer)
    __builtin_prefetch(arow0 + kb + 32, 0, 0);
    __builtin_prefetch(arow1 + kb + 32, 0, 0);

    v16bf a0 = load_frag(arow0, kb);
    v16bf a1 = load_frag(arow1, kb);

    v16bf b;
    b = load_frag(brow0, kb);
    acc[0] = __builtin_amdgcn_wmma_f32_16x16x32_bf16(false, a0, false, b, (short)0, acc[0], false, false);
    acc[4] = __builtin_amdgcn_wmma_f32_16x16x32_bf16(false, a1, false, b, (short)0, acc[4], false, false);
    b = load_frag(brow1, kb);
    acc[1] = __builtin_amdgcn_wmma_f32_16x16x32_bf16(false, a0, false, b, (short)0, acc[1], false, false);
    acc[5] = __builtin_amdgcn_wmma_f32_16x16x32_bf16(false, a1, false, b, (short)0, acc[5], false, false);
    b = load_frag(brow2, kb);
    acc[2] = __builtin_amdgcn_wmma_f32_16x16x32_bf16(false, a0, false, b, (short)0, acc[2], false, false);
    acc[6] = __builtin_amdgcn_wmma_f32_16x16x32_bf16(false, a1, false, b, (short)0, acc[6], false, false);
    b = load_frag(brow3, kb);
    acc[3] = __builtin_amdgcn_wmma_f32_16x16x32_bf16(false, a0, false, b, (short)0, acc[3], false, false);
    acc[7] = __builtin_amdgcn_wmma_f32_16x16x32_bf16(false, a1, false, b, (short)0, acc[7], false, false);
  }

  // epilogue: VGPR r holds M = r + 8*hl (+16 for i=1), N = l16 (per C/D layout)
  #pragma unroll
  for (int i = 0; i < 2; ++i) {
    #pragma unroll
    for (int s = 0; s < 4; ++s) {
      const int col = n0 + s * 16 + l16;
      const float bv = bias ? bias[col] : 0.0f;
      #pragma unroll
      for (int r = 0; r < 8; ++r) {
        const int row = m0 + i * 16 + hl * 8 + r;
        float v = acc[i * 4 + s][r] + bv;
        if (act == 1)      v = v > 0.0f ? v : 0.0f;               // relu
        else if (act == 2) v = v > 0.0f ? v : (__expf(v) - 1.0f); // elu
        if (C)   C[(size_t)row * N + col] = v;
        if (Cbf) Cbf[(size_t)row * N + col] = (bf16_t)v;
      }
    }
  }
}

// ---------------- ef assembly: f32 to d_out + padded bf16 for WMMA ---------
__global__ void k_ef(const float* __restrict__ emb, const int* __restrict__ bc,
                     const float* __restrict__ tval, const float* __restrict__ isroot,
                     float* __restrict__ ef_out, bf16_t* __restrict__ ef_b) {
  const int u = blockIdx.x;  // 0..8191 (8191 = zero pad row)
  bf16_t* eb = ef_b + (size_t)u * KEP;
  if (u < KNN) {
    const float* hf = emb + (size_t)KNN * KHID;
    const float* ht = emb + (size_t)bc[u] * KHID;
    float* er = ef_out + (size_t)u * KE;
    for (int ci = threadIdx.x; ci < KHID; ci += blockDim.x) {
      float a = hf[ci], b = ht[ci];
      float ad = fabsf(a - b), p = a * b;
      er[ci] = a; er[KHID + ci] = b; er[2 * KHID + ci] = ad; er[3 * KHID + ci] = p;
      eb[ci] = (bf16_t)a; eb[KHID + ci] = (bf16_t)b;
      eb[2 * KHID + ci] = (bf16_t)ad; eb[3 * KHID + ci] = (bf16_t)p;
    }
    if (threadIdx.x == 0) {
      float tn = tval[u] * (1.0f / (1.0f + 1e-8f));
      er[2048] = tn; er[2049] = isroot[u];
      eb[2048] = (bf16_t)tn; eb[2049] = (bf16_t)isroot[u];
      for (int j = 2050; j < KEP; ++j) eb[j] = (bf16_t)0.0f;
    }
  } else {
    for (int ci = threadIdx.x; ci < KEP; ci += blockDim.x) eb[ci] = (bf16_t)0.0f;
  }
}

// ---------------- logits matvec: dot(z2[u], W3) + b3 -----------------------
__global__ __launch_bounds__(256) void k_logits(const float* __restrict__ z2,
                                                const float* __restrict__ W3,
                                                const float* __restrict__ b3,
                                                float* __restrict__ logits) {
  __shared__ float red[256];
  const int u = blockIdx.x;
  float s = 0.0f;
  for (int ci = threadIdx.x; ci < KHID; ci += 256)
    s += z2[(size_t)u * KHID + ci] * W3[ci];
  red[threadIdx.x] = s; __syncthreads();
  for (int off = 128; off > 0; off >>= 1) {
    if (threadIdx.x < off) red[threadIdx.x] += red[threadIdx.x + off];
    __syncthreads();
  }
  if (threadIdx.x == 0) logits[u] = red[0] + b3[0];
}

// ---------------- one-block softmax over 8191 logits -----------------------
__global__ __launch_bounds__(1024) void k_softmax(const float* __restrict__ logits,
                                                  float* __restrict__ probs, int n) {
  __shared__ float red[1024];
  __shared__ float smax, ssum;
  float m = -3.4e38f;
  for (int i = threadIdx.x; i < n; i += 1024) m = fmaxf(m, logits[i]);
  red[threadIdx.x] = m; __syncthreads();
  for (int off = 512; off > 0; off >>= 1) {
    if (threadIdx.x < off) red[threadIdx.x] = fmaxf(red[threadIdx.x], red[threadIdx.x + off]);
    __syncthreads();
  }
  if (threadIdx.x == 0) smax = red[0];
  __syncthreads();
  float s = 0.0f;
  for (int i = threadIdx.x; i < n; i += 1024) s += __expf(logits[i] - smax);
  red[threadIdx.x] = s; __syncthreads();
  for (int off = 512; off > 0; off >>= 1) {
    if (threadIdx.x < off) red[threadIdx.x] += red[threadIdx.x + off];
    __syncthreads();
  }
  if (threadIdx.x == 0) ssum = red[0];
  __syncthreads();
  for (int i = threadIdx.x; i < n; i += 1024) probs[i] = __expf(logits[i] - smax) / ssum;
}

// ---------------- focal one-hot output -------------------------------------
__global__ void k_focal(const int* __restrict__ focal, float* __restrict__ out) {
  const int i = blockIdx.x * blockDim.x + threadIdx.x;
  if (i < KNF) out[i] = (i == focal[0]) ? 1.0f : 0.0f;
}

// =================== launcher ===============================================
extern "C" void kernel_launch(void* const* d_in, const int* in_sizes, int n_in,
                              void* d_out, int out_size, void* d_ws, size_t ws_size,
                              hipStream_t stream) {
  (void)in_sizes; (void)n_in; (void)out_size; (void)ws_size;
  const int*   ns     = (const int*)d_in[0];
  const int*   ch0    = (const int*)d_in[1];
  const int*   ch1    = (const int*)d_in[2];
  const int*   parent = (const int*)d_in[3];
  const int*   neigh  = (const int*)d_in[4];
  const int*   bc     = (const int*)d_in[5];
  const int*   focal  = (const int*)d_in[6];
  const float* tval   = (const float*)d_in[7];
  const float* isroot = (const float*)d_in[8];
  const float* W1  = (const float*)d_in[9];   const float* b1  = (const float*)d_in[10];
  const float* W2  = (const float*)d_in[11];  const float* b2  = (const float*)d_in[12];
  const float* Wh1 = (const float*)d_in[13];  const float* bh1 = (const float*)d_in[14];
  const float* Wh2 = (const float*)d_in[15];  const float* bh2 = (const float*)d_in[16];
  const float* W3  = (const float*)d_in[17];  const float* b3  = (const float*)d_in[18];

  // ---- workspace layout (manual carve, 256B aligned) ----
  char* ws = (char*)d_ws;
  float*  c_buf = (float*)(ws + 0x0);
  float*  d_buf = (float*)(ws + 0x10000);                        // 8192x4096 f32 = 0x8000000
  bf16_t* agg1  = (bf16_t*)(ws + 0x10000 + 0x8000000);           // 8192x4096 bf16 = 0x4000000
  bf16_t* w1t   = (bf16_t*)(ws + 0xC010000);                     // 512x4096 bf16 = 0x400000
  bf16_t* w2t   = (bf16_t*)(ws + 0xC410000);                     // 512x512 bf16  = 0x80000
  bf16_t* wh1t  = (bf16_t*)(ws + 0xC490000);                     // 512x2080 bf16 = 0x208000
  bf16_t* wh2t  = (bf16_t*)(ws + 0xC698000);                     // 512x512 bf16  = 0x80000
  float*  h_f32 = (float*)(ws + 0xC718000);                      // 8192x512 f32  = 0x1000000
  bf16_t* agg2  = (bf16_t*)(ws + 0xD718000);                     // 8192x512 bf16 = 0x800000
  // reuse the d_buf region once GCN1 has consumed it:
  bf16_t* efb   = (bf16_t*)(ws + 0x10000);                       // 8192x2080 bf16 = 0x2080000
  bf16_t* z1b   = (bf16_t*)(ws + 0x10000 + 0x2080000);           // 8192x512 bf16
  float*  z2f   = (float*)(ws + 0x10000 + 0x2080000 + 0x800000); // 8192x512 f32

  // ---- output layout (flat f32, concatenated in return order) ----
  float* out_logits = (float*)d_out;
  float* out_probs  = out_logits + KNN;
  float* out_ef     = out_probs + KNN;
  float* out_emb    = out_ef + (size_t)KNN * KE;
  float* out_focal  = out_emb + (size_t)KROWS * KHID;

  // 1) scalar c recursion + column-parallel belief prop
  k_cvals<<<1, 32, 0, stream>>>(ns, ch0, ch1, c_buf);
  k_belief<<<KNF / 32, 32, 0, stream>>>(ns, ch0, ch1, parent, c_buf, focal, d_buf);

  // 2) weight prep (f32 KxN -> bf16 NxKpad transposed)
  k_wt<<<dim3(KNF / 256, KHID), 256, 0, stream>>>(W1, w1t, KNF, KHID, KNF);
  k_wt<<<dim3(KHID / 256, KHID), 256, 0, stream>>>(W2, w2t, KHID, KHID, KHID);
  k_wt<<<dim3((KEP + 255) / 256, KHID), 256, 0, stream>>>(Wh1, wh1t, KE, KHID, KEP);
  k_wt<<<dim3(KHID / 256, KHID), 256, 0, stream>>>(Wh2, wh2t, KHID, KHID, KHID);

  // 3) GCN layer 1: h = relu(agg(d) @ W1 + b1)
  k_agg<<<dim3(KNF / 256, KROWS), 256, 0, stream>>>(d_buf, neigh, agg1, KNF);
  k_gemm_wmma<<<dim3(KROWS / 128, KHID / 64), 128, 0, stream>>>(
      agg1, w1t, b1, h_f32, nullptr, KHID, KNF, /*relu*/1);

  // 4) GCN layer 2: emb = agg(h) @ W2 + b2  (written straight to d_out)
  k_agg<<<dim3(KHID / 256, KROWS), 256, 0, stream>>>(h_f32, neigh, agg2, KHID);
  k_gemm_wmma<<<dim3(KROWS / 128, KHID / 64), 128, 0, stream>>>(
      agg2, w2t, b2, out_emb, nullptr, KHID, KHID, /*none*/0);

  // 5) pairwise head features (f32 to out, padded bf16 for WMMA)
  k_ef<<<KROWS, 256, 0, stream>>>(out_emb, bc, tval, isroot, out_ef, efb);

  // 6) head MLP: z1 = elu(ef @ Wh1 + bh1); z2 = elu(z1 @ Wh2 + bh2)
  k_gemm_wmma<<<dim3(KROWS / 128, KHID / 64), 128, 0, stream>>>(
      efb, wh1t, bh1, nullptr, z1b, KHID, KEP, /*elu*/2);
  k_gemm_wmma<<<dim3(KROWS / 128, KHID / 64), 128, 0, stream>>>(
      z1b, wh2t, bh2, z2f, nullptr, KHID, KHID, /*elu*/2);

  // 7) logits, softmax, focal one-hot
  k_logits<<<KNN, 256, 0, stream>>>(z2f, W3, b3, out_logits);
  k_softmax<<<1, 1024, 0, stream>>>(out_logits, out_probs, KNN);
  k_focal<<<KNF / 256, 256, 0, stream>>>(focal, out_focal);
}
